// MultiHeadAttention_45457933861305
// MI455X (gfx1250) — compile-verified, matrix operand order checked by reading, not run on hardware
//
#include <hip/hip_runtime.h>
#include <hip/hip_bf16.h>

typedef __attribute__((ext_vector_type(16))) __bf16 v16bf;
typedef __attribute__((ext_vector_type(8)))  __bf16 v8bf;
typedef __attribute__((ext_vector_type(8)))  float  v8f;

#define B_  4
#define L_  1024
#define U_  256
#define H_  8
#define DH_ 32
#define TD_ 8

__device__ __forceinline__ float relu_(float x){ return x > 0.f ? x : 0.f; }

// -------------------------------------------------------------------------
// 1) time features: per (b,l) run three 1->8->8 relu MLPs on the timestamp.
// feats layout per row (stride 32 floats): [0..7]=2*pi*period*t, [8..15]=sigma+1e-6,
// [16..23]=basis, [24]=t, [25..31]=0
// -------------------------------------------------------------------------
__device__ void ffn8_(float tv, const float* W1, const float* b1,
                      const float* W2, const float* b2, float* out) {
  float h[TD_];
#pragma unroll
  for (int j = 0; j < TD_; ++j) h[j] = relu_(tv * W1[j] + b1[j]);
#pragma unroll
  for (int j = 0; j < TD_; ++j) {
    float s = b2[j];
#pragma unroll
    for (int i = 0; i < TD_; ++i) s += h[i] * W2[i * TD_ + j];
    out[j] = relu_(s);
  }
}

__global__ void time_feat_kernel(const float* __restrict__ t,
    const float* pW1, const float* pb1, const float* pW2, const float* pb2,
    const float* sW1, const float* sb1, const float* sW2, const float* sb2,
    const float* bW1, const float* bb1, const float* bW2, const float* bb2,
    float* __restrict__ feats) {
  int id = blockIdx.x * blockDim.x + threadIdx.x;
  if (id >= B_ * L_) return;
  float tv = t[id];
  float* f = feats + (size_t)id * 32;
  float tmp[TD_];
  ffn8_(tv, pW1, pb1, pW2, pb2, tmp);
#pragma unroll
  for (int j = 0; j < TD_; ++j) f[j] = 6.283185307179586f * tmp[j] * tv;
  ffn8_(tv, sW1, sb1, sW2, sb2, tmp);
#pragma unroll
  for (int j = 0; j < TD_; ++j) f[8 + j] = tmp[j] + 1e-6f;
  ffn8_(tv, bW1, bb1, bW2, bb2, tmp);
#pragma unroll
  for (int j = 0; j < TD_; ++j) f[16 + j] = tmp[j];
  f[24] = tv;
#pragma unroll
  for (int j = 25; j < 32; ++j) f[j] = 0.f;
}

// -------------------------------------------------------------------------
// 2) conversions: activations fp32->bf16, weights fp32 -> transposed bf16
// -------------------------------------------------------------------------
__global__ void cvt_bf16_kernel(const float* __restrict__ src,
                                __bf16* __restrict__ dst, int n) {
  int i = blockIdx.x * blockDim.x + threadIdx.x;
  int stride = gridDim.x * blockDim.x;
  for (; i < n; i += stride) dst[i] = (__bf16)src[i];
}

__global__ void wt_bf16_kernel(const float* __restrict__ W,
                               __bf16* __restrict__ WT) {
  int i = blockIdx.x * blockDim.x + threadIdx.x;  // 65536 threads
  int n = i >> 8, k = i & 255;
  WT[n * 256 + k] = (__bf16)W[k * 256 + n];
}

// -------------------------------------------------------------------------
// 3) QKV projection GEMM: one 16x16 output tile per wave, K=256 in 8 WMMA.
//    A[M,K] bf16 row-major, W^T[N,K] bf16 row-major.
//    bf16 A/B fragment lane layout (16x16x32): lanes<16 hold K {0..7,16..23},
//    lanes>=16 hold K {8..15,24..31}; row/col = lane&15.
// -------------------------------------------------------------------------
__global__ __launch_bounds__(128) void qkv_gemm_kernel(
    const __bf16* __restrict__ xq, const __bf16* __restrict__ xi,
    const __bf16* __restrict__ wqt, const __bf16* __restrict__ wkt,
    const __bf16* __restrict__ wvt,
    __bf16* __restrict__ Qb, __bf16* __restrict__ Kb, __bf16* __restrict__ Vtb) {
  int wave = threadIdx.x >> 5;
  int lane = threadIdx.x & 31;
  int gw = blockIdx.x * 4 + wave;            // 0 .. 12287
  int p = gw / 4096;                         // 0=Q, 1=K, 2=V
  int rem = gw - p * 4096;
  int m0 = (rem >> 4) * 16;
  int n0 = (rem & 15) * 16;
  const __bf16* A = (p == 0) ? xq : xi;
  const __bf16* W = (p == 0) ? wqt : (p == 1 ? wkt : wvt);
  int lh = lane >> 4;                        // half index
  int lrow = lane & 15;
  int kb = lh * 8;                           // K base within 16
  const __bf16* pA = A + (size_t)(m0 + lrow) * U_;
  const __bf16* pW = W + (size_t)(n0 + lrow) * U_;
  v8f c = {};
#pragma unroll
  for (int kk = 0; kk < U_; kk += 32) {
    v8bf a0 = *reinterpret_cast<const v8bf*>(pA + kk + kb);
    v8bf a1 = *reinterpret_cast<const v8bf*>(pA + kk + 16 + kb);
    v8bf b0 = *reinterpret_cast<const v8bf*>(pW + kk + kb);
    v8bf b1 = *reinterpret_cast<const v8bf*>(pW + kk + 16 + kb);
    v16bf af, bf;
#pragma unroll
    for (int i = 0; i < 8; ++i) {
      af[i] = a0[i]; af[i + 8] = a1[i];
      bf[i] = b0[i]; bf[i + 8] = b1[i];
    }
    c = __builtin_amdgcn_wmma_f32_16x16x32_bf16(false, af, false, bf,
                                                (short)0, c, false, false);
  }
  // C layout: VGPR r -> row m0+r (lanes<16) / m0+8+r (lanes>=16), col = lane&15
#pragma unroll
  for (int r = 0; r < 8; ++r) {
    int m = m0 + r + lh * 8;
    int n = n0 + lrow;
    int b = m >> 10, l = m & 1023;
    int h = n >> 5, d = n & 31;
    __bf16 v = (__bf16)c[r];
    if (p == 0)      Qb [(((size_t)(b * H_) + h) * L_ + l) * DH_ + d] = v;
    else if (p == 1) Kb [(((size_t)(b * H_) + h) * L_ + l) * DH_ + d] = v;
    else             Vtb[(((size_t)(b * H_) + h) * DH_ + d) * L_ + l] = v;
  }
}

// -------------------------------------------------------------------------
// 4) key / query sign masks per (b,h,l)
// -------------------------------------------------------------------------
__global__ void mask_kernel(const __bf16* __restrict__ Qb,
                            const __bf16* __restrict__ Kb,
                            float* __restrict__ qmask, float* __restrict__ kmask) {
  int i = blockIdx.x * blockDim.x + threadIdx.x;
  if (i >= B_ * H_ * L_) return;
  float sq = 0.f, sk = 0.f;
#pragma unroll
  for (int d = 0; d < DH_; ++d) {
    sq += fabsf((float)Qb[(size_t)i * DH_ + d]);
    sk += fabsf((float)Kb[(size_t)i * DH_ + d]);
  }
  qmask[i] = (sq != 0.f) ? 1.f : 0.f;
  kmask[i] = (sk != 0.f) ? 1.f : 0.f;
}

// -------------------------------------------------------------------------
// 5) time-kernel tk[B,L,L], computed once (shared by all heads), LDS-tiled
// -------------------------------------------------------------------------
__global__ __launch_bounds__(256) void tk_kernel(const float* __restrict__ feats,
                                                 float* __restrict__ tkbuf) {
  __shared__ float fq[16][32];
  __shared__ float fk[16][32];
  int b = blockIdx.z;
  int k0 = blockIdx.x * 16;
  int q0 = blockIdx.y * 16;
  int tx = threadIdx.x, ty = threadIdx.y;
  int tid = ty * 16 + tx;
  for (int i = tid; i < 16 * 32; i += 256) {
    int row = i >> 5, c = i & 31;
    fq[row][c] = feats[((size_t)(b * L_ + q0 + row)) * 32 + c];
    fk[row][c] = feats[((size_t)(b * L_ + k0 + row)) * 32 + c];
  }
  __syncthreads();
  const float* q = fq[ty];
  const float* k = fk[tx];
  float dt = q[24] - k[24];
  float dt2 = dt * dt;
  float acc = 0.f;
#pragma unroll
  for (int d = 0; d < TD_; ++d) {
    float cosv = __cosf(q[d] - k[d]);
    float a = q[8 + d], bb = k[8 + d];
    float inv = 1.0f / (a * a + bb * bb);
    float e = __expf(-dt2 * inv);
    float loc = sqrtf(2.f * a * bb * inv);
    acc += q[16 + d] * k[16 + d] * loc * e * cosv;
  }
  tkbuf[((size_t)(b * L_ + q0 + ty)) * L_ + (k0 + tx)] = acc;
}

// -------------------------------------------------------------------------
// 6) fused flash attention: wave = 16 query rows of one (b,h).
//    Per 32-key tile: 2 WMMA for S = Q*K^T (dh=32 == full K depth),
//    online softmax (shfl reductions within 16-lane halves = C-layout rows),
//    P staged through a WAVE-PRIVATE LDS tile (C-layout -> A-layout + bf16),
//    2 WMMA for O += P*V using V^T layout.
//    No block barriers needed: LDS ops of one wave complete in order (DScnt),
//    and each wave only touches its own LDS region -> waves run untethered.
// -------------------------------------------------------------------------
__global__ __launch_bounds__(128) void attn_kernel(
    const __bf16* __restrict__ Qb, const __bf16* __restrict__ Kb,
    const __bf16* __restrict__ Vtb, const float* __restrict__ tkbuf,
    const float* __restrict__ kmask, const float* __restrict__ qmask,
    const int* __restrict__ inmask, float* __restrict__ out) {
  __shared__ __align__(16) float Pld[4][16][32];
  int wave = threadIdx.x >> 5;
  int lane = threadIdx.x & 31;
  int bh = blockIdx.x >> 4;       // 0..31
  int qblk = blockIdx.x & 15;
  int b = bh >> 3, h = bh & 7;
  int lh = lane >> 4;
  int lrow = lane & 15;
  int kb = lh * 8;
  int q0 = qblk * 64 + wave * 16;

  const float scale = 0.17677669529663687f;   // 1/sqrt(32)
  const float NEGB = -4294967295.0f;          // -2^32 + 1

  // Q fragment: rows q0..q0+15, K-dim = all 32 head dims (loaded once)
  v16bf qf;
  {
    const __bf16* pQ = Qb + ((size_t)bh * L_ + (q0 + lrow)) * DH_;
    v8bf a0 = *reinterpret_cast<const v8bf*>(pQ + kb);
    v8bf a1 = *reinterpret_cast<const v8bf*>(pQ + 16 + kb);
#pragma unroll
    for (int i = 0; i < 8; ++i) { qf[i] = a0[i]; qf[i + 8] = a1[i]; }
  }

  float m[8], lsum[8];
  v8f o0 = {}, o1 = {};
#pragma unroll
  for (int r = 0; r < 8; ++r) { m[r] = -3.0e38f; lsum[r] = 0.f; }

  for (int kt = 0; kt < L_; kt += 32) {
    // ---- prefetch next tile (global_prefetch_b8): overlap HBM/L2 latency
    //      with this tile's softmax VALU chain
    if (kt + 32 < L_) {
      __builtin_prefetch(Kb + ((size_t)bh * L_ + (kt + 32 + lane)) * DH_, 0, 1);
      __builtin_prefetch(Vtb + ((size_t)bh * DH_ + lane) * L_ + kt + 32, 0, 1);
      __builtin_prefetch(tkbuf + ((size_t)(b * L_ + q0 + lrow)) * L_ + kt + 32, 0, 1);
    }

    // ---- scores: two 16-key sub-tiles
    v8f s0, s1;
    {
      const __bf16* pK = Kb + ((size_t)bh * L_ + (kt + lrow)) * DH_;
      v8bf b0 = *reinterpret_cast<const v8bf*>(pK + kb);
      v8bf b1 = *reinterpret_cast<const v8bf*>(pK + 16 + kb);
      v16bf kf;
#pragma unroll
      for (int i = 0; i < 8; ++i) { kf[i] = b0[i]; kf[i + 8] = b1[i]; }
      v8f z = {};
      s0 = __builtin_amdgcn_wmma_f32_16x16x32_bf16(false, qf, false, kf,
                                                   (short)0, z, false, false);
    }
    {
      const __bf16* pK = Kb + ((size_t)bh * L_ + (kt + 16 + lrow)) * DH_;
      v8bf b0 = *reinterpret_cast<const v8bf*>(pK + kb);
      v8bf b1 = *reinterpret_cast<const v8bf*>(pK + 16 + kb);
      v16bf kf;
#pragma unroll
      for (int i = 0; i < 8; ++i) { kf[i] = b0[i]; kf[i + 8] = b1[i]; }
      v8f z = {};
      s1 = __builtin_amdgcn_wmma_f32_16x16x32_bf16(false, qf, false, kf,
                                                   (short)0, z, false, false);
    }

    // ---- bias + masks + online softmax (per C-fragment row r)
    float p0[8], p1[8];
#pragma unroll
    for (int r = 0; r < 8; ++r) {
      int q = q0 + r + lh * 8;
      int k0i = kt + lrow;
      int k1i = kt + 16 + lrow;
      float v0 = (s0[r] + tkbuf[((size_t)(b * L_ + q)) * L_ + k0i]) * scale;
      float v1 = (s1[r] + tkbuf[((size_t)(b * L_ + q)) * L_ + k1i]) * scale;
      if (kmask[(size_t)bh * L_ + k0i] == 0.f) v0 = NEGB;
      if (kmask[(size_t)bh * L_ + k1i] == 0.f) v1 = NEGB;
      if (q <= k0i) v0 -= 10000.f;   // strict causal: allowed only q > k
      if (q <= k1i) v1 -= 10000.f;
      float rv = fmaxf(v0, v1);
#pragma unroll
      for (int off = 8; off >= 1; off >>= 1)
        rv = fmaxf(rv, __shfl_xor(rv, off, 32));
      float mn = fmaxf(m[r], rv);
      float corr = __expf(m[r] - mn);
      float e0 = __expf(v0 - mn);
      float e1 = __expf(v1 - mn);
      float rs = e0 + e1;
#pragma unroll
      for (int off = 8; off >= 1; off >>= 1) rs += __shfl_xor(rs, off, 32);
      lsum[r] = lsum[r] * corr + rs;
      m[r] = mn;
      o0[r] *= corr;
      o1[r] *= corr;
      p0[r] = e0;
      p1[r] = e1;
    }

    // ---- stage P (C layout) into wave-private LDS tile; same-wave LDS is
    //      in-order (DScnt), compiler inserts the store->load wait.
#pragma unroll
    for (int r = 0; r < 8; ++r) {
      Pld[wave][r + lh * 8][lrow]      = p0[r];
      Pld[wave][r + lh * 8][16 + lrow] = p1[r];
    }

    // ---- rebuild P as bf16 A-fragment (rows = q, K = 32 keys)
    v16bf pf;
    {
      const float4* pr = reinterpret_cast<const float4*>(&Pld[wave][lrow][0]);
      float4 x0 = pr[(kb >> 2) + 0];
      float4 x1 = pr[(kb >> 2) + 1];
      float4 x2 = pr[((16 + kb) >> 2) + 0];
      float4 x3 = pr[((16 + kb) >> 2) + 1];
      pf[0] = (__bf16)x0.x;  pf[1] = (__bf16)x0.y;
      pf[2] = (__bf16)x0.z;  pf[3] = (__bf16)x0.w;
      pf[4] = (__bf16)x1.x;  pf[5] = (__bf16)x1.y;
      pf[6] = (__bf16)x1.z;  pf[7] = (__bf16)x1.w;
      pf[8] = (__bf16)x2.x;  pf[9] = (__bf16)x2.y;
      pf[10] = (__bf16)x2.z; pf[11] = (__bf16)x2.w;
      pf[12] = (__bf16)x3.x; pf[13] = (__bf16)x3.y;
      pf[14] = (__bf16)x3.z; pf[15] = (__bf16)x3.w;
    }

    // ---- O += P * V  (V stored transposed [B,H,dh,L] -> contiguous keys)
    {
      const __bf16* pV0 = Vtb + ((size_t)bh * DH_ + lrow) * L_ + kt;
      const __bf16* pV1 = Vtb + ((size_t)bh * DH_ + 16 + lrow) * L_ + kt;
      v8bf c00 = *reinterpret_cast<const v8bf*>(pV0 + kb);
      v8bf c01 = *reinterpret_cast<const v8bf*>(pV0 + 16 + kb);
      v8bf c10 = *reinterpret_cast<const v8bf*>(pV1 + kb);
      v8bf c11 = *reinterpret_cast<const v8bf*>(pV1 + 16 + kb);
      v16bf vf0, vf1;
#pragma unroll
      for (int i = 0; i < 8; ++i) {
        vf0[i] = c00[i]; vf0[i + 8] = c01[i];
        vf1[i] = c10[i]; vf1[i + 8] = c11[i];
      }
      o0 = __builtin_amdgcn_wmma_f32_16x16x32_bf16(false, pf, false, vf0,
                                                   (short)0, o0, false, false);
      o1 = __builtin_amdgcn_wmma_f32_16x16x32_bf16(false, pf, false, vf1,
                                                   (short)0, o1, false, false);
    }
  }

  // ---- epilogue: 1/l, query mask, input mask, store [B,L,H*dh]
#pragma unroll
  for (int r = 0; r < 8; ++r) {
    int q = q0 + r + lh * 8;
    float qm = qmask[(size_t)bh * L_ + q];
    float im = (float)inmask[b * L_ + q];
    float inv = 1.0f / lsum[r];
    float f0 = o0[r] * inv * qm * im;
    float f1 = o1[r] * inv * qm * im;
    size_t base = ((size_t)(b * L_ + q)) * U_ + h * DH_;
    out[base + lrow] = f0;
    out[base + 16 + lrow] = f1;
  }
}

// -------------------------------------------------------------------------
extern "C" void kernel_launch(void* const* d_in, const int* in_sizes, int n_in,
                              void* d_out, int out_size, void* d_ws, size_t ws_size,
                              hipStream_t stream) {
  const float* query  = (const float*)d_in[0];
  const float* inputt = (const float*)d_in[1];
  const int*   inmask = (const int*)d_in[2];
  const float* timei  = (const float*)d_in[3];
  // d_in[4] = att_bias (unused by reference math)
  const float* Wq = (const float*)d_in[5];
  const float* Wk = (const float*)d_in[6];
  const float* Wv = (const float*)d_in[7];
  const float* pW1 = (const float*)d_in[8];
  const float* pb1 = (const float*)d_in[9];
  const float* pW2 = (const float*)d_in[10];
  const float* pb2 = (const float*)d_in[11];
  const float* sW1 = (const float*)d_in[12];
  const float* sb1 = (const float*)d_in[13];
  const float* sW2 = (const float*)d_in[14];
  const float* sb2 = (const float*)d_in[15];
  const float* bW1 = (const float*)d_in[16];
  const float* bb1 = (const float*)d_in[17];
  const float* bW2 = (const float*)d_in[18];
  const float* bb2 = (const float*)d_in[19];

  char* ws = (char*)d_ws;
  size_t off = 0;
  auto alloc = [&](size_t bytes) -> char* {
    char* p = ws + off;
    off += (bytes + 255) & ~(size_t)255;
    return p;
  };
  float*  feats  = (float*) alloc((size_t)B_ * L_ * 32 * sizeof(float));
  float*  tkbuf  = (float*) alloc((size_t)B_ * L_ * L_ * sizeof(float));
  __bf16* xq     = (__bf16*)alloc((size_t)B_ * L_ * U_ * sizeof(__bf16));
  __bf16* xi     = (__bf16*)alloc((size_t)B_ * L_ * U_ * sizeof(__bf16));
  __bf16* wqt    = (__bf16*)alloc((size_t)U_ * U_ * sizeof(__bf16));
  __bf16* wkt    = (__bf16*)alloc((size_t)U_ * U_ * sizeof(__bf16));
  __bf16* wvt    = (__bf16*)alloc((size_t)U_ * U_ * sizeof(__bf16));
  __bf16* Qb     = (__bf16*)alloc((size_t)B_ * H_ * L_ * DH_ * sizeof(__bf16));
  __bf16* Kb     = (__bf16*)alloc((size_t)B_ * H_ * L_ * DH_ * sizeof(__bf16));
  __bf16* Vtb    = (__bf16*)alloc((size_t)B_ * H_ * DH_ * L_ * sizeof(__bf16));
  float*  qmaskp = (float*) alloc((size_t)B_ * H_ * L_ * sizeof(float));
  float*  kmaskp = (float*) alloc((size_t)B_ * H_ * L_ * sizeof(float));
  if (off > ws_size) return;  // workspace too small; nothing safe to do

  time_feat_kernel<<<(B_ * L_ + 255) / 256, 256, 0, stream>>>(
      timei, pW1, pb1, pW2, pb2, sW1, sb1, sW2, sb2, bW1, bb1, bW2, bb2, feats);

  cvt_bf16_kernel<<<512, 256, 0, stream>>>(query,  xq, B_ * L_ * U_);
  cvt_bf16_kernel<<<512, 256, 0, stream>>>(inputt, xi, B_ * L_ * U_);
  wt_bf16_kernel<<<256, 256, 0, stream>>>(Wq, wqt);
  wt_bf16_kernel<<<256, 256, 0, stream>>>(Wk, wkt);
  wt_bf16_kernel<<<256, 256, 0, stream>>>(Wv, wvt);

  qkv_gemm_kernel<<<3072, 128, 0, stream>>>(xq, xi, wqt, wkt, wvt, Qb, Kb, Vtb);

  mask_kernel<<<(B_ * H_ * L_ + 255) / 256, 256, 0, stream>>>(Qb, Kb, qmaskp, kmaskp);

  dim3 tgrid(L_ / 16, L_ / 16, B_);
  tk_kernel<<<tgrid, dim3(16, 16), 0, stream>>>(feats, tkbuf);

  attn_kernel<<<B_ * H_ * 16, 128, 0, stream>>>(
      Qb, Kb, Vtb, tkbuf, kmaskp, qmaskp, inmask, (float*)d_out);
}
